// GNN_53919019434191
// MI455X (gfx1250) — matrix operand order repaired and checked by reading, not compiled, not run on hardware
//
#include <hip/hip_runtime.h>
#include <math.h>

// ---------------------------------------------------------------------------
// GNN pipeline for MI455X (gfx1250, wave32).
// Heavy math (the three SAGE projections) runs on v_wmma_f32_16x16x32_f16.
// A-tiles are staged with global_load_async_to_lds_b128 (double-buffered,
// s_wait_asynccnt pipelined); B weights are pre-swizzled into WMMA fragment
// order with the column normalization folded into the first-layer weights.
// ---------------------------------------------------------------------------

typedef __attribute__((ext_vector_type(16))) _Float16 v16h;
typedef __attribute__((ext_vector_type(8)))  float    v8f;

#define N_NODES   131072
#define XDIM      1036
#define HDIM      128
#define FDIM      32
#define NEDGE     1048576
#define NGRAPH    64
#define NPAT      16

union HFrag { _Float16 h[16]; unsigned u[8]; v16h v; };

// ---------------------------------------------------------------------------
// 0) zero helper
// ---------------------------------------------------------------------------
__global__ void k_zero(float* __restrict__ p, long n) {
  long i = (long)blockIdx.x * blockDim.x + threadIdx.x;
  long stride = (long)gridDim.x * blockDim.x;
  for (; i < n; i += stride) p[i] = 0.0f;
}

// ---------------------------------------------------------------------------
// 1) column max of first 12 cols of x; stores RECIPROCALS (divide -> mul).
// ---------------------------------------------------------------------------
__global__ __launch_bounds__(1024) void k_colmax(const float* __restrict__ x,
                                                 float* __restrict__ rcmax) {
  __shared__ float red[1024];
  const int tid = threadIdx.x;
  float lm[12];
#pragma unroll
  for (int c = 0; c < 12; ++c) lm[c] = -3.0e38f;
  for (int r = tid; r < N_NODES; r += 1024) {
    const float* row = x + (size_t)r * XDIM;
#pragma unroll
    for (int c = 0; c < 12; ++c) lm[c] = fmaxf(lm[c], row[c]);
  }
  for (int c = 0; c < 12; ++c) {
    red[tid] = lm[c];
    __syncthreads();
    for (int s = 512; s > 0; s >>= 1) {
      if (tid < s) red[tid] = fmaxf(red[tid], red[tid + s]);
      __syncthreads();
    }
    if (tid == 0) rcmax[c] = 1.0f / red[0];
    __syncthreads();
  }
}

// ---------------------------------------------------------------------------
// 2) Pre-swizzle fused weights B = [Wl | Wr]  (K x 256, fp32) into WMMA
//    B-fragment order (f16).  x[:, :12]/cmax @ W == x @ (rows<12 of W scaled
//    by 1/cmax), so the input normalization is folded in here (normN=12 for
//    layer 0 only).  Fragment layout used here (contiguous k, same order as A):
//      lane L: n_local = L%16; element e of v16h <-> k_local = (L/16)*16 + e;
//      flat frag index = (kt*16 + nt)*32 + lane.
// ---------------------------------------------------------------------------
__global__ void k_prep_wfrag(const float* __restrict__ Wl,
                             const float* __restrict__ Wr,
                             v16h* __restrict__ out, int K, int KT,
                             int normN, const float* __restrict__ rcmax) {
  int t = blockIdx.x * blockDim.x + threadIdx.x;
  if (t >= KT * 512) return;
  int kt   = t >> 9;
  int rem  = t & 511;
  int nt   = rem >> 5;
  int lane = rem & 31;
  int khalf = lane >> 4;
  int n = nt * 16 + (lane & 15);
  HFrag hf;
#pragma unroll
  for (int e = 0; e < 16; ++e) {
    int k = kt * 32 + khalf * 16 + e;
    float w = 0.0f;
    if (k < K) {
      w = (n < HDIM) ? Wl[(size_t)k * HDIM + n]
                     : Wr[(size_t)k * HDIM + (n - HDIM)];
      if (k < normN) w *= rcmax[k];
    }
    hf.h[e] = (_Float16)w;
  }
  out[t] = hf.v;  // t == (kt*16+nt)*32 + lane
}

// ---------------------------------------------------------------------------
// 3) async A-tile staging: copy one 64x32 fp32 tile (row stride padded to 36
//    floats = 144B, 16B-aligned) global -> LDS via the gfx1250 async path.
//    Each thread moves 2 x 16B granules (K and lda are multiples of 4, so
//    granules are fully valid or fully OOB; OOB granules are zero-filled).
//    Two async instructions per wave per tile -> s_wait_asynccnt 0x2 retires
//    the previous tile while the next is in flight (in-order completion).
// ---------------------------------------------------------------------------
__device__ __forceinline__ void stage_tile_async(
    const float* __restrict__ A, int lda, int K, int row0, int kt,
    float* bufBase /* LDS */, int tid) {
  const int r  = tid >> 2;
  const int q0 = (tid & 3) * 2;
  float* rowp = bufBase + r * 36;
#pragma unroll
  for (int qq = 0; qq < 2; ++qq) {
    int q  = q0 + qq;
    int k0 = kt * 32 + q * 4;
    float* lp = rowp + q * 4;
    if (k0 < K) {
      const float* gp = A + (size_t)(row0 + r) * lda + k0;
      unsigned loff = (unsigned)(uintptr_t)lp;  // low 32 bits = LDS offset
      asm volatile("global_load_async_to_lds_b128 %0, %1, off"
                   :: "v"(loff), "v"(gp) : "memory");
    } else {
      lp[0] = 0.f; lp[1] = 0.f; lp[2] = 0.f; lp[3] = 0.f;
    }
  }
}

// ---------------------------------------------------------------------------
// 4) WMMA GEMM:  C[M x 256] = A[M x K](fp32) @ Wfrag(K x 256, f16 fragments)
//    Block: 256 thr = 8 waves; tile 64 rows x 256 cols per block.
//    Wave (rt,ch): rows rt*16..+16, cols ch*128..+128 (8 accumulators v8f).
//    All 8 B fragments are preloaded into registers before the WMMA chain so
//    the matrix pipe issues back-to-back behind a single wait.
//    All M values (131072 / 26240 / 5248) are multiples of 64.
// ---------------------------------------------------------------------------
__global__ __launch_bounds__(256) void k_gemm_wmma(
    const float* __restrict__ A, const v16h* __restrict__ Wfrag,
    float* __restrict__ C, int lda, int K, int KT) {
  __shared__ __attribute__((aligned(16))) float Abuf[2][64 * 36];
  const int tid  = threadIdx.x;
  const int row0 = blockIdx.x * 64;
  const int wid  = tid >> 5, lane = tid & 31;
  const int rt = wid >> 1, ch = wid & 1;
  const int ml   = rt * 16 + (lane & 15);
  const int half = lane >> 4;

  v8f acc[8];
#pragma unroll
  for (int j = 0; j < 8; ++j) {
    v8f z = {0.f, 0.f, 0.f, 0.f, 0.f, 0.f, 0.f, 0.f};
    acc[j] = z;
  }

  stage_tile_async(A, lda, K, row0, 0, Abuf[0], tid);

  for (int kt = 0; kt < KT; ++kt) {
    if (kt + 1 < KT) {
      stage_tile_async(A, lda, K, row0, kt + 1, Abuf[(kt + 1) & 1], tid);
      asm volatile("s_wait_asynccnt 0x2" ::: "memory");
    } else {
      asm volatile("s_wait_asynccnt 0x0" ::: "memory");
    }
    __syncthreads();  // tile kt fully in LDS (all waves)

    // preload all 8 B fragments (each: 32B = 2 x b128 from L2-resident wf)
    const v16h* wp = Wfrag + ((size_t)(kt * 16 + ch * 8) * 32 + lane);
    v16h bfr[8];
#pragma unroll
    for (int j = 0; j < 8; ++j) bfr[j] = wp[(size_t)j * 32];

    // A fragment: VGPR v holds K pair k0,k0+1 with
    //   k0 = v*2 + (lane/16)*16   (contiguous order, matches Wfrag B layout)
    const float* bp = &Abuf[kt & 1][ml * 36];
    HFrag af;
#pragma unroll
    for (int v2 = 0; v2 < 8; ++v2) {
      int k0 = (half << 4) + (v2 << 1);
      float f0 = bp[k0], f1 = bp[k0 + 1];
      union { _Float16 h[2]; unsigned u; } pk;
      pk.h[0] = (_Float16)f0;
      pk.h[1] = (_Float16)f1;
      af.u[v2] = pk.u;
    }

#pragma unroll
    for (int j = 0; j < 8; ++j) {
      acc[j] = __builtin_amdgcn_wmma_f32_16x16x32_f16(
          false, af.v, false, bfr[j], (short)0, acc[j], false, false);
    }
    __syncthreads();  // all waves done reading Abuf[kt&1] before overwrite
  }

  // C fragment: lane n = L%16, VGPR v -> row m = v + 8*(L/16)
#pragma unroll
  for (int j = 0; j < 8; ++j) {
    int col = (ch * 8 + j) * 16 + (lane & 15);
#pragma unroll
    for (int v2 = 0; v2 < 8; ++v2) {
      int row = row0 + rt * 16 + half * 8 + v2;
      C[(size_t)row * 256 + col] = acc[j][v2];
    }
  }
}

// ---------------------------------------------------------------------------
// 5) initial edge materialization (valid = 1 everywhere)
// ---------------------------------------------------------------------------
__global__ void k_init_edges(const int* __restrict__ ei,
                             int* __restrict__ es, int* __restrict__ ed,
                             int* __restrict__ ev) {
  int e = blockIdx.x * blockDim.x + threadIdx.x;
  if (e >= NEDGE) return;
  es[e] = ei[e];
  ed[e] = ei[NEDGE + e];
  ev[e] = 1;
}

// ---------------------------------------------------------------------------
// 6) edge aggregation: AGG[dst] += hl[src], DEG[dst] += 1 (masked).
//    One 32-lane wavefront per edge; channels across lanes (coalesced,
//    gathered hl region fits in 192MB L2).
// ---------------------------------------------------------------------------
__global__ __launch_bounds__(256) void k_edge_agg(
    const int* __restrict__ es, const int* __restrict__ ed,
    const int* __restrict__ ev, const float* __restrict__ C,
    float* __restrict__ AGG, float* __restrict__ DEG) {
  const int tid = threadIdx.x;
  const int lane = tid & 31;
  const int e = blockIdx.x * 8 + (tid >> 5);
  if (!ev[e]) return;
  const int s = es[e], d = ed[e];
  const float* hl = C + (size_t)s * 256;
  float* ag = AGG + (size_t)d * 128;
#pragma unroll
  for (int q = 0; q < 4; ++q) {
    int c = lane + q * 32;
    atomicAdd(&ag[c], hl[c]);
  }
  if (lane == 0) atomicAdd(&DEG[d], 1.0f);
}

// ---------------------------------------------------------------------------
// 7) finalize SAGE:  H = relu(AGG/max(deg,1) + bl + hr), written into AGG.
// ---------------------------------------------------------------------------
__global__ void k_finalize(float* __restrict__ AGG, const float* __restrict__ DEG,
                           const float* __restrict__ C,
                           const float* __restrict__ bl, int M) {
  long idx = (long)blockIdx.x * blockDim.x + threadIdx.x;
  if (idx >= (long)M * 128) return;
  int i = (int)(idx >> 7), c = (int)(idx & 127);
  float d = fmaxf(DEG[i], 1.0f);
  float h = AGG[idx] / d + bl[c] + C[(size_t)i * 256 + 128 + c];
  AGG[idx] = fmaxf(h, 0.0f);
}

// ---------------------------------------------------------------------------
// 8) SAGPool scores.  PS[i] = H[i].Prel ; SC[i] = H[i].Proot + prelb
//    then SC[dst] += PS[src] over valid edges.
// ---------------------------------------------------------------------------
__global__ void k_score_nodes(const float* __restrict__ H,
                              const float* __restrict__ Prel,
                              const float* __restrict__ Proot,
                              const float* __restrict__ prelb,
                              float* __restrict__ PS, float* __restrict__ SC,
                              int M) {
  int i = blockIdx.x * blockDim.x + threadIdx.x;
  if (i >= M) return;
  const float* h = H + (size_t)i * 128;
  float s1 = 0.f, s2 = 0.f;
  for (int c = 0; c < 128; ++c) {
    s1 += h[c] * Prel[c];
    s2 += h[c] * Proot[c];
  }
  PS[i] = s1;
  SC[i] = s2 + prelb[0];
}

__global__ void k_score_edges(const int* __restrict__ es,
                              const int* __restrict__ ed,
                              const int* __restrict__ ev,
                              const float* __restrict__ PS,
                              float* __restrict__ SC) {
  int e = blockIdx.x * blockDim.x + threadIdx.x;
  if (e >= NEDGE || !ev[e]) return;
  atomicAdd(&SC[ed[e]], PS[es[e]]);
}

// ---------------------------------------------------------------------------
// 9) per-graph top-k pool (bitonic sort in LDS, descending) + gate + mapping.
//    Downstream readouts are permutation-invariant, so selecting the same
//    top-k set is equivalent to jax.lax.top_k.
// ---------------------------------------------------------------------------
__global__ __launch_bounds__(256) void k_pool(
    const float* __restrict__ H, const float* __restrict__ SC,
    float* __restrict__ XPout, int* __restrict__ MAP,
    int npg, int npgPad, int kkeep) {
  __shared__ float ss[2048];
  __shared__ int   si[2048];
  const int g = blockIdx.x, tid = threadIdx.x;
  for (int i = tid; i < npgPad; i += 256) {
    ss[i] = (i < npg) ? SC[(size_t)g * npg + i] : -3.0e38f;
    si[i] = i;
  }
  __syncthreads();
  for (int size = 2; size <= npgPad; size <<= 1) {
    for (int stride = size >> 1; stride > 0; stride >>= 1) {
      for (int i = tid; i < npgPad; i += 256) {
        int j = i ^ stride;
        if (j > i) {
          bool low = (i & size) == 0;  // descending overall
          bool sw = low ? (ss[i] < ss[j]) : (ss[i] > ss[j]);
          if (sw) {
            float tf = ss[i]; ss[i] = ss[j]; ss[j] = tf;
            int   ti = si[i]; si[i] = si[j]; si[j] = ti;
          }
        }
      }
      __syncthreads();
    }
  }
  for (int i = tid; i < npg; i += 256) MAP[(size_t)g * npg + i] = -1;
  __syncthreads();
  for (int t = tid; t < kkeep; t += 256)
    MAP[(size_t)g * npg + si[t]] = g * kkeep + t;
  for (int idx = tid; idx < kkeep * 128; idx += 256) {
    int t = idx >> 7, c = idx & 127;
    float gate = tanhf(ss[t]);
    XPout[(size_t)(g * kkeep + t) * 128 + c] =
        H[(size_t)(g * npg + si[t]) * 128 + c] * gate;
  }
}

__global__ void k_remap(const int* __restrict__ es, const int* __restrict__ ed,
                        const int* __restrict__ ev, const int* __restrict__ MAP,
                        int* __restrict__ ns, int* __restrict__ nd,
                        int* __restrict__ nv) {
  int e = blockIdx.x * blockDim.x + threadIdx.x;
  if (e >= NEDGE) return;
  int a = MAP[es[e]], b = MAP[ed[e]];
  int v = ev[e] && a >= 0 && b >= 0;
  ns[e] = a >= 0 ? a : 0;
  nd[e] = b >= 0 ? b : 0;
  nv[e] = v;
}

// ---------------------------------------------------------------------------
// 10) readout: XS[g][c] += max, XS[g][128+c] += mean over pooled nodes.
// ---------------------------------------------------------------------------
__global__ void k_readout(const float* __restrict__ XP, float* __restrict__ XS,
                          int kkeep) {
  const int g = blockIdx.x, c = threadIdx.x;  // 128 threads
  float m = -3.0e38f, s = 0.f;
  for (int t = 0; t < kkeep; ++t) {
    float v = XP[(size_t)(g * kkeep + t) * 128 + c];
    m = fmaxf(m, v);
    s += v;
  }
  XS[(size_t)g * 256 + c]       += m;
  XS[(size_t)g * 256 + 128 + c] += s / (float)kkeep;
}

// ---------------------------------------------------------------------------
// 11) head MLPs (tiny)
// ---------------------------------------------------------------------------
__global__ void k_mlp1(const float* __restrict__ XS, const float* __restrict__ W,
                       const float* __restrict__ b, float* __restrict__ HG) {
  const int g = blockIdx.x, j = threadIdx.x;  // 128 threads
  float a = b[j];
  for (int c = 0; c < 256; ++c) a += XS[(size_t)g * 256 + c] * W[(size_t)c * 128 + j];
  HG[(size_t)g * 128 + j] = fmaxf(a, 0.f);
}

__global__ void k_mlp2(const float* __restrict__ HG, const float* __restrict__ W,
                       const float* __restrict__ b, float* __restrict__ F) {
  const int g = blockIdx.x, f = threadIdx.x;  // 32 threads
  float a = b[f];
  for (int c = 0; c < 128; ++c) a += HG[(size_t)g * 128 + c] * W[(size_t)c * 32 + f];
  F[(size_t)g * 32 + f] = fmaxf(a, 0.f);
}

// ---------------------------------------------------------------------------
// 12) gated attention pool + grade/hazard.  One block.  Writes all 576 outs:
//     [0,512) xp, [512,560) grade, [560,576) hazard.
// ---------------------------------------------------------------------------
__global__ __launch_bounds__(256) void k_head(
    const float* __restrict__ F, const int* __restrict__ pat,
    const float* __restrict__ Wa, const float* __restrict__ ba,
    const float* __restrict__ Wb, const float* __restrict__ bb,
    const float* __restrict__ Wc, const float* __restrict__ bc,
    const float* __restrict__ Wg, const float* __restrict__ bg,
    const float* __restrict__ Wh, const float* __restrict__ bh,
    float* __restrict__ out) {
  __shared__ float As[NGRAPH], Wts[NGRAPH], Gm[NPAT], Gs[NPAT], XPl[NPAT * 32];
  const int tid = threadIdx.x;
  if (tid < NGRAPH) {
    const float* f = F + (size_t)tid * 32;
    float a = 0.f;
    for (int t = 0; t < 16; ++t) {
      float sa = ba[t], sb = bb[t];
      for (int c = 0; c < 32; ++c) {
        sa += f[c] * Wa[c * 16 + t];
        sb += f[c] * Wb[c * 16 + t];
      }
      a += (1.0f / (1.0f + expf(-sa))) * tanhf(sb) * Wc[t];
    }
    As[tid] = a + bc[0];
  }
  __syncthreads();
  if (tid < NPAT) {
    float m = -3.0e38f;
    for (int g = 0; g < NGRAPH; ++g)
      if (pat[g] == tid) m = fmaxf(m, As[g]);
    Gm[tid] = m;
  }
  __syncthreads();
  if (tid < NGRAPH) Wts[tid] = expf(As[tid] - Gm[pat[tid]]);
  __syncthreads();
  if (tid < NPAT) {
    float s = 0.f;
    for (int g = 0; g < NGRAPH; ++g)
      if (pat[g] == tid) s += Wts[g];
    Gs[tid] = s;
  }
  __syncthreads();
  for (int idx = tid; idx < NPAT * 32; idx += 256) {
    int p = idx >> 5, c = idx & 31;
    float v = 0.f;
    for (int g = 0; g < NGRAPH; ++g)
      if (pat[g] == p) v += (Wts[g] / Gs[p]) * F[(size_t)g * 32 + c];
    XPl[idx] = v;
    out[idx] = v;
  }
  __syncthreads();
  if (tid < NPAT) {
    const float* xp = XPl + tid * 32;
    float l[3];
    float m = -3.0e38f;
    for (int c = 0; c < 3; ++c) {
      float a = bg[c];
      for (int f2 = 0; f2 < 32; ++f2) a += xp[f2] * Wg[f2 * 3 + c];
      l[c] = a;
      m = fmaxf(m, a);
    }
    float se = 0.f;
    for (int c = 0; c < 3; ++c) se += expf(l[c] - m);
    float lse = m + logf(se);
    for (int c = 0; c < 3; ++c) out[512 + tid * 3 + c] = l[c] - lse;
    float h = bh[0];
    for (int f2 = 0; f2 < 32; ++f2) h += xp[f2] * Wh[f2];
    out[560 + tid] = 6.0f / (1.0f + expf(-h)) - 3.0f;
  }
}

// ---------------------------------------------------------------------------
// host-side orchestration
// ---------------------------------------------------------------------------
extern "C" void kernel_launch(void* const* d_in, const int* in_sizes, int n_in,
                              void* d_out, int out_size, void* d_ws,
                              size_t ws_size, hipStream_t stream) {
  (void)in_sizes; (void)n_in; (void)out_size; (void)ws_size;

  // inputs in setup_inputs() dict (insertion) order
  const float* x  = (const float*)d_in[0];
  const int*   ei = (const int*)d_in[1];
  const int*   pat = (const int*)d_in[3];  // d_in[2] = batch (unused)
  const int PB = 4;  // params: Wl0,bl0,Wr0,Prel0,prelb0,Proot0, x3 layers,
                     // We1,be1,We2,be2,Wa,ba,Wb,bb,Wc,bc,Wg,bg,Wh,bh
  const float* Wl[3]    = {(const float*)d_in[PB + 0], (const float*)d_in[PB + 6],  (const float*)d_in[PB + 12]};
  const float* bl[3]    = {(const float*)d_in[PB + 1], (const float*)d_in[PB + 7],  (const float*)d_in[PB + 13]};
  const float* Wr[3]    = {(const float*)d_in[PB + 2], (const float*)d_in[PB + 8],  (const float*)d_in[PB + 14]};
  const float* Prel[3]  = {(const float*)d_in[PB + 3], (const float*)d_in[PB + 9],  (const float*)d_in[PB + 15]};
  const float* prelb[3] = {(const float*)d_in[PB + 4], (const float*)d_in[PB + 10], (const float*)d_in[PB + 16]};
  const float* Proot[3] = {(const float*)d_in[PB + 5], (const float*)d_in[PB + 11], (const float*)d_in[PB + 17]};
  const float* We1 = (const float*)d_in[PB + 18];
  const float* be1 = (const float*)d_in[PB + 19];
  const float* We2 = (const float*)d_in[PB + 20];
  const float* be2 = (const float*)d_in[PB + 21];
  const float* Wa  = (const float*)d_in[PB + 22];
  const float* ba  = (const float*)d_in[PB + 23];
  const float* Wb  = (const float*)d_in[PB + 24];
  const float* bb  = (const float*)d_in[PB + 25];
  const float* Wc  = (const float*)d_in[PB + 26];
  const float* bc  = (const float*)d_in[PB + 27];
  const float* Wg  = (const float*)d_in[PB + 28];
  const float* bg  = (const float*)d_in[PB + 29];
  const float* Wh  = (const float*)d_in[PB + 30];
  const float* bh  = (const float*)d_in[PB + 31];
  float* out = (float*)d_out;

  // workspace carve-up (~256 MB total)
  char* base = (char*)d_ws;
  size_t off = 0;
  auto alloc = [&](size_t bytes) -> void* {
    off = (off + 255) & ~(size_t)255;
    void* p = base + off;
    off += bytes;
    return p;
  };
  float* rcmax = (float*)alloc(64 * 4);
  v16h*  wf0   = (v16h*)alloc((size_t)33 * 512 * 32);
  v16h*  wf1   = (v16h*)alloc((size_t)4 * 512 * 32);
  v16h*  wf2   = (v16h*)alloc((size_t)4 * 512 * 32);
  float* C     = (float*)alloc((size_t)N_NODES * 256 * 4);
  float* AGG   = (float*)alloc((size_t)N_NODES * 128 * 4);  // doubles as H
  float* DEG   = (float*)alloc((size_t)N_NODES * 4);
  float* PS    = (float*)alloc((size_t)N_NODES * 4);
  float* SC    = (float*)alloc((size_t)N_NODES * 4);
  int*   MAP   = (int*)alloc((size_t)N_NODES * 4);
  float* XPa   = (float*)alloc((size_t)26240 * 128 * 4);
  float* XPb   = (float*)alloc((size_t)26240 * 128 * 4);
  int*   es0   = (int*)alloc((size_t)NEDGE * 4);
  int*   ed0   = (int*)alloc((size_t)NEDGE * 4);
  int*   ev0   = (int*)alloc((size_t)NEDGE * 4);
  int*   es1   = (int*)alloc((size_t)NEDGE * 4);
  int*   ed1   = (int*)alloc((size_t)NEDGE * 4);
  int*   ev1   = (int*)alloc((size_t)NEDGE * 4);
  float* XS    = (float*)alloc((size_t)NGRAPH * 256 * 4);
  float* HG    = (float*)alloc((size_t)NGRAPH * 128 * 4);
  float* FF    = (float*)alloc((size_t)NGRAPH * 32 * 4);

  // preprocessing (colmax BEFORE weight swizzle: norm is folded into wf0)
  k_colmax<<<1, 1024, 0, stream>>>(x, rcmax);
  k_zero<<<64, 256, 0, stream>>>(XS, (long)NGRAPH * 256);
  k_prep_wfrag<<<(33 * 512 + 255) / 256, 256, 0, stream>>>(Wl[0], Wr[0], wf0, XDIM, 33, 12, rcmax);
  k_prep_wfrag<<<(4 * 512 + 255) / 256, 256, 0, stream>>>(Wl[1], Wr[1], wf1, HDIM, 4, 0, rcmax);
  k_prep_wfrag<<<(4 * 512 + 255) / 256, 256, 0, stream>>>(Wl[2], Wr[2], wf2, HDIM, 4, 0, rcmax);
  k_init_edges<<<NEDGE / 256, 256, 0, stream>>>(ei, es0, ed0, ev0);

  const int   Ms[3]    = {N_NODES, 64 * 410, 64 * 82};
  const int   Ks[3]    = {XDIM, HDIM, HDIM};
  const int   KTs[3]   = {33, 4, 4};
  const int   npgs[3]  = {2048, 410, 82};
  const int   pads[3]  = {2048, 512, 128};
  const int   keeps[3] = {410, 82, 17};
  const v16h* wfs[3]   = {wf0, wf1, wf2};
  const float* Ains[3] = {x, XPa, XPb};
  float*       XPo[3]  = {XPa, XPb, XPa};
  int* ces[3] = {es0, es1, es0}; int* ced[3] = {ed0, ed1, ed0}; int* cev[3] = {ev0, ev1, ev0};
  int* nes[3] = {es1, es0, 0};   int* ned[3] = {ed1, ed0, 0};   int* nev[3] = {ev1, ev0, 0};

  for (int L = 0; L < 3; ++L) {
    const int M = Ms[L];
    k_gemm_wmma<<<M / 64, 256, 0, stream>>>(Ains[L], wfs[L], C, Ks[L], Ks[L], KTs[L]);
    k_zero<<<1024, 256, 0, stream>>>(AGG, (long)M * 128);
    k_zero<<<64, 256, 0, stream>>>(DEG, (long)M);
    k_edge_agg<<<NEDGE / 8, 256, 0, stream>>>(ces[L], ced[L], cev[L], C, AGG, DEG);
    k_finalize<<<(M * 128) / 256, 256, 0, stream>>>(AGG, DEG, C, bl[L], M);
    k_score_nodes<<<(M + 255) / 256, 256, 0, stream>>>(AGG, Prel[L], Proot[L],
                                                       prelb[L], PS, SC, M);
    k_score_edges<<<NEDGE / 256, 256, 0, stream>>>(ces[L], ced[L], cev[L], PS, SC);
    k_pool<<<NGRAPH, 256, 0, stream>>>(AGG, SC, XPo[L], MAP, npgs[L], pads[L], keeps[L]);
    if (L < 2)
      k_remap<<<NEDGE / 256, 256, 0, stream>>>(ces[L], ced[L], cev[L], MAP,
                                               nes[L], ned[L], nev[L]);
    k_readout<<<NGRAPH, 128, 0, stream>>>(XPo[L], XS, keeps[L]);
  }

  k_mlp1<<<NGRAPH, 128, 0, stream>>>(XS, We1, be1, HG);
  k_mlp2<<<NGRAPH, 32, 0, stream>>>(HG, We2, be2, FF);
  k_head<<<1, 256, 0, stream>>>(FF, pat, Wa, ba, Wb, bb, Wc, bc, Wg, bg, Wh, bh, out);
}
